// Node_1829656068286
// MI455X (gfx1250) — matrix operand order repaired and checked by reading, not compiled, not run on hardware
//
#include <hip/hip_runtime.h>
#include <math.h>

typedef __attribute__((ext_vector_type(16))) _Float16 v16h;
typedef __attribute__((ext_vector_type(8)))  float    v8f;

#define HID    128
#define NP     64            // pixels per block tile
#define LSTR   136           // padded LDS row stride in halves (136*2B = 272B, conflict-free)
#define NPIX   (256*512)

union FragB { uint4 u[2]; v16h h; };
union Pack8 { uint4 u; _Float16 h[8]; };
union PackF4 { float4 v; float f[4]; };

// Branch-free fast tanh: 1 - 2/(exp2(x*2*log2e)+1).
// Exact at +/-inf (no inf/inf NaN), ~1e-6 rel error, 5 VALU ops (2 trans).
__device__ __forceinline__ float tanh_fast(float x) {
  float e = __builtin_amdgcn_exp2f(x * 2.8853900817779268f);
  return fmaf(-2.0f, __builtin_amdgcn_rcpf(e + 1.0f), 1.0f);
}

// One hidden GEMM layer: out = tanh(resp*(W@in + r*prev) + bias).
// FIRST==true skips the recurrent term entirely (compile-time).
template <bool FIRST>
__device__ __forceinline__ void gemm_layer(
    const v16h* __restrict__ A,          // 4 K-chunk A fragments (wave's 16 rows)
    const _Float16* __restrict__ bin,    // input activations  [NP][LSTR]
    const _Float16* __restrict__ prev,   // recurrent state    [NP][LSTR]
    _Float16* __restrict__ bdst,         // output activations [NP][LSTR]
    const float* __restrict__ rl,
    const float* __restrict__ respl,
    const float* __restrict__ biasl,
    int wave, int lane)
{
  const int colBase = lane & 15;
  const int kbB     = (lane & 16) ? 8 : 0;    // B frag: lanes 0-15 -> K 0..7,16..23, lanes 16-31 -> K 8..15,24..31
  const int rowBase = 16 * wave + ((lane & 16) ? 8 : 0);  // C frag: VGPR r -> M = rowBase + r

  // Preload this lane's 8 rows of per-channel params once (2x b128 LDS loads each).
  PackF4 r01, r23, s01, s23, b01, b23;
  if (!FIRST) {
    r01.v = *(const float4*)(rl + rowBase);
    r23.v = *(const float4*)(rl + rowBase + 4);
  }
  s01.v = *(const float4*)(respl + rowBase);
  s23.v = *(const float4*)(respl + rowBase + 4);
  b01.v = *(const float4*)(biasl + rowBase);
  b23.v = *(const float4*)(biasl + rowBase + 4);

  for (int n = 0; n < NP / 16; ++n) {
    const int col = n * 16 + colBase;
    v8f c = {0.f, 0.f, 0.f, 0.f, 0.f, 0.f, 0.f, 0.f};
#pragma unroll
    for (int ch = 0; ch < 4; ++ch) {
      FragB fb;
      const uint4* src = (const uint4*)(bin + col * LSTR + ch * 32 + kbB);
      fb.u[0] = src[0];                  // halves K kb+0..7
      fb.u[1] = src[2];                  // halves K kb+16..23
      c = __builtin_amdgcn_wmma_f32_16x16x32_f16(
              false, A[ch], false, fb.h, (short)0, c, false, false);
    }
    // Epilogue: recurrent add + response/bias/tanh, pack 8 halves, one b128 LDS store.
    Pack8 pv, st;
    if (!FIRST) pv.u = *(const uint4*)(prev + col * LSTR + rowBase);
#pragma unroll
    for (int r = 0; r < 8; ++r) {
      const float rr = (r < 4) ? r01.f[r & 3] : r23.f[r & 3];
      const float ss = (r < 4) ? s01.f[r & 3] : s23.f[r & 3];
      const float bb = (r < 4) ? b01.f[r & 3] : b23.f[r & 3];
      float pre = c[r];
      if (!FIRST) pre = fmaf(rr, (float)pv.h[r], pre);
      st.h[r] = (_Float16)tanh_fast(fmaf(ss, pre, bb));
    }
    *(uint4*)(bdst + col * LSTR + rowBase) = st.u;
  }
}

__global__ __launch_bounds__(256) void rec_dag_kernel(
    const float* __restrict__ x_in,      // [3][NPIX]
    const float* __restrict__ W_in,      // [128][3]
    const float* __restrict__ W_h,       // [2][128][128]
    const float* __restrict__ r_rec,     // [3][128]
    const float* __restrict__ resp,      // [3][128]
    const float* __restrict__ bias,      // [3][128]
    const float* __restrict__ W_out,     // [3][128]
    const float* __restrict__ resp_out,  // [3]
    const float* __restrict__ b_out,     // [3]
    const int*   __restrict__ n_iters_p,
    float* __restrict__ out)             // [3][NPIX]
{
  __shared__ _Float16 hbuf[4][NP * LSTR];     // rotating activation buffers (f16)
  __shared__ float xs[3 * NP];
  __shared__ float Win_s[HID * 3];
  __shared__ float r_s[3 * HID], resp_s[3 * HID], bias_s[3 * HID], Wout_s[3 * HID];

  const int tid  = threadIdx.x;
  const int lane = tid & 31;
  const int wave = tid >> 5;
  const int pix0 = blockIdx.x * NP;
  const int n_iters = *n_iters_p;

  // ---- stage small params + x tile into LDS ----
  for (int i = tid; i < HID * 3; i += 256) Win_s[i] = W_in[i];
  for (int i = tid; i < 3 * HID; i += 256) {
    r_s[i] = r_rec[i]; resp_s[i] = resp[i]; bias_s[i] = bias[i]; Wout_s[i] = W_out[i];
  }
  for (int i = tid; i < 3 * NP; i += 256) {
    const int c = i / NP, p = i % NP;
    xs[c * NP + p] = x_in[c * NPIX + pix0 + p];
  }

  // ---- preload this wave's A fragments for W_h[0] and W_h[1] (f32 -> f16) ----
  // A 16x32 f16 layout: lanes 0-15 row M=lane hold K kb..kb+7 (elems 0-7) and
  // kb+16..kb+23 (elems 8-15) with kb=0; lanes 16-31 same rows with kb=8.
  v16h a0[4], a1[4];
  {
    const int row  = 16 * wave + (lane & 15);
    const int koff = (lane & 16) ? 8 : 0;
#pragma unroll
    for (int m = 0; m < 2; ++m) {
      const float* W = W_h + m * HID * HID + row * HID;
#pragma unroll
      for (int ch = 0; ch < 4; ++ch) {
        const int kb = ch * 32 + koff;
        const float4 f0 = *(const float4*)(W + kb);
        const float4 f1 = *(const float4*)(W + kb + 4);
        const float4 f2 = *(const float4*)(W + kb + 16);
        const float4 f3 = *(const float4*)(W + kb + 20);
        v16h a;
        a[0]  = (_Float16)f0.x; a[1]  = (_Float16)f0.y; a[2]  = (_Float16)f0.z; a[3]  = (_Float16)f0.w;
        a[4]  = (_Float16)f1.x; a[5]  = (_Float16)f1.y; a[6]  = (_Float16)f1.z; a[7]  = (_Float16)f1.w;
        a[8]  = (_Float16)f2.x; a[9]  = (_Float16)f2.y; a[10] = (_Float16)f2.z; a[11] = (_Float16)f2.w;
        a[12] = (_Float16)f3.x; a[13] = (_Float16)f3.y; a[14] = (_Float16)f3.z; a[15] = (_Float16)f3.w;
        if (m == 0) a0[ch] = a; else a1[ch] = a;
      }
    }
  }
  __syncthreads();

  // Rotating buffer indices: p0/p1/p2 hold previous h0/h1/h2, fr is free.
  int p0 = 0, p1 = 1, p2 = 2, fr = 3;

  for (int t = 0; t < n_iters; ++t) {
    const bool first = (t == 0);

    // ---- layer 0 (leaf, 3-wide): h0_new -> hbuf[fr] ----
    if (first) {
      for (int i = tid; i < NP * HID; i += 256) {
        const int p = i >> 7, h = i & 127;
        float pre = Win_s[h * 3 + 0] * xs[0 * NP + p]
                  + Win_s[h * 3 + 1] * xs[1 * NP + p]
                  + Win_s[h * 3 + 2] * xs[2 * NP + p];
        hbuf[fr][p * LSTR + h] = (_Float16)tanh_fast(fmaf(resp_s[h], pre, bias_s[h]));
      }
    } else {
      for (int i = tid; i < NP * HID; i += 256) {
        const int p = i >> 7, h = i & 127;
        float pre = Win_s[h * 3 + 0] * xs[0 * NP + p]
                  + Win_s[h * 3 + 1] * xs[1 * NP + p]
                  + Win_s[h * 3 + 2] * xs[2 * NP + p];
        pre = fmaf(r_s[h], (float)hbuf[p0][p * LSTR + h], pre);
        hbuf[fr][p * LSTR + h] = (_Float16)tanh_fast(fmaf(resp_s[h], pre, bias_s[h]));
      }
    }
    __syncthreads();
    const int n0 = fr;

    // ---- layer 1: Wh0 @ h0_new (+ r1*h1_prev) -> hbuf[p0] (h0_prev now dead) ----
    if (first) {
      gemm_layer<true>(a0, hbuf[n0], hbuf[p1], hbuf[p0],
                       r_s + HID, resp_s + HID, bias_s + HID, wave, lane);
    } else {
      gemm_layer<false>(a0, hbuf[n0], hbuf[p1], hbuf[p0],
                        r_s + HID, resp_s + HID, bias_s + HID, wave, lane);
    }
    __syncthreads();
    const int n1 = p0;

    // ---- layer 2: Wh1 @ h1_new (+ r2*h2_prev) -> hbuf[p1] (h1_prev now dead) ----
    if (first) {
      gemm_layer<true>(a1, hbuf[n1], hbuf[p2], hbuf[p1],
                       r_s + 2 * HID, resp_s + 2 * HID, bias_s + 2 * HID, wave, lane);
    } else {
      gemm_layer<false>(a1, hbuf[n1], hbuf[p2], hbuf[p1],
                        r_s + 2 * HID, resp_s + 2 * HID, bias_s + 2 * HID, wave, lane);
    }
    __syncthreads();
    const int n2 = p1;

    // rotate: new prev = (n0, n1, n2); free = old h2_prev slot
    const int nf = p2;
    p0 = n0; p1 = n1; p2 = n2; fr = nf;
  }

  // ---- output layer (3-wide) from final h2 in hbuf[p2] ----
  for (int i = tid; i < 3 * NP; i += 256) {
    const int o = i / NP, p = i % NP;
    const _Float16* hp = &hbuf[p2][p * LSTR];
    const float* w = &Wout_s[o * HID];
    float s = 0.f;
#pragma unroll 8
    for (int h = 0; h < HID; ++h) s = fmaf(w[h], (float)hp[h], s);
    out[o * NPIX + pix0 + p] = tanh_fast(fmaf(resp_out[o], s, b_out[o]));
  }
}

extern "C" void kernel_launch(void* const* d_in, const int* in_sizes, int n_in,
                              void* d_out, int out_size, void* d_ws, size_t ws_size,
                              hipStream_t stream) {
  (void)in_sizes; (void)n_in; (void)out_size; (void)d_ws; (void)ws_size;
  const float* x_in     = (const float*)d_in[0];
  const float* W_in     = (const float*)d_in[1];
  const float* W_h      = (const float*)d_in[2];
  const float* r_rec    = (const float*)d_in[3];
  const float* resp     = (const float*)d_in[4];
  const float* bias     = (const float*)d_in[5];
  const float* W_out    = (const float*)d_in[6];
  const float* resp_out = (const float*)d_in[7];
  const float* b_out    = (const float*)d_in[8];
  const int*   n_iters  = (const int*)d_in[9];
  float*       out      = (float*)d_out;

  const int nblocks = NPIX / NP;   // 2048 tiles of 64 pixels
  rec_dag_kernel<<<dim3(nblocks), dim3(256), 0, stream>>>(
      x_in, W_in, W_h, r_rec, resp, bias, W_out, resp_out, b_out, n_iters, out);
}
